// STTemporalAttention_81303730913776
// MI455X (gfx1250) — matrix operand order repaired and checked by reading, not compiled, not run on hardware
//
#include <hip/hip_runtime.h>
#include <cstdint>
#include <cstddef>

typedef unsigned short ushort_t;
typedef __attribute__((ext_vector_type(16))) __bf16 v16bf;
typedef __attribute__((ext_vector_type(8)))  float  v8f;
typedef __attribute__((ext_vector_type(4)))  unsigned int u32x4;
typedef __attribute__((ext_vector_type(8)))  int i32x8;
typedef __attribute__((ext_vector_type(4)))  int i32x4;

union Frag { uint4 u4[2]; v16bf v; };

#if defined(__has_builtin)
#  if __has_builtin(__builtin_amdgcn_tensor_load_to_lds) && __has_builtin(__builtin_amdgcn_s_wait_tensorcnt)
#    define USE_TDM 1
#  endif
#endif
#ifndef USE_TDM
#  define USE_TDM 0
#endif

__device__ __forceinline__ ushort_t f32_bf16(float f) {
    union { float f; unsigned int u; } c;
    c.f = f;
    unsigned int r = (c.u + 0x7FFFu + ((c.u >> 16) & 1u)) >> 16;
    return (ushort_t)r;
}

// ---------------------------------------------------------------- pack fp32 -> bf16
__global__ void cvt_bf16_kernel(const float* __restrict__ in, ushort_t* __restrict__ out, int n) {
    int i = blockIdx.x * blockDim.x + threadIdx.x;
    if (i < n) out[i] = f32_bf16(in[i]);
}

// ---------------------------------------------------------------- TDM tile DMA
// Loads a (128 rows x 32 elements) bf16 tile, row stride K elements, into LDS
// with 16B of padding after each 64B row => LDS row stride 80B (LDSW=40 elems).
#if USE_TDM
__device__ __forceinline__ void tdm_load_tile(unsigned lds_addr, const ushort_t* gaddr, int K) {
    const unsigned long long ga = (unsigned long long)(uintptr_t)gaddr;
    u32x4 g0;
    g0[0] = 1u;                                   // count=1, is_restore=0, gather off
    g0[1] = lds_addr;                             // LDS byte address
    g0[2] = (unsigned)(ga & 0xFFFFFFFFu);         // global_addr[31:0]
    g0[3] = (unsigned)((ga >> 32) & 0x01FFFFFFu)  // global_addr[56:32]
            | (2u << 30);                         // type=2 (image)
    i32x8 g1;
    g1[0] = (1 << 16)     // data_size = 1 -> 2 bytes
          | (1 << 20)     // pad_enable
          | (3 << 22)     // pad_interval: 1<<3 = 8 x 8B = 64B
          | (3 << 25);    // pad_amount: 4 DWORDs = 16B
    g1[1] = (int)((K & 0xFFFF) << 16);                            // tensor_dim0[15:0]
    g1[2] = (int)((((unsigned)K >> 16) & 0xFFFFu) | (128u << 16)); // dim0[31:16] | tensor_dim1[15:0]=128
    g1[3] = (int)(32u << 16);                                     // tile_dim0 = 32 elements
    g1[4] = 128;                                                  // tile_dim1 = 128 rows, tile_dim2=0
    g1[5] = K;                                                    // tensor_dim0_stride[31:0] (elements)
    g1[6] = 0;                                                    // stride[47:32]=0, dim1_stride=0
    g1[7] = 0;
    i32x4 gz = {0, 0, 0, 0};
#if __clang_major__ >= 23
    i32x8 gz8 = {0, 0, 0, 0, 0, 0, 0, 0};
    __builtin_amdgcn_tensor_load_to_lds(g0, g1, gz, gz, gz8, 0);
#else
    __builtin_amdgcn_tensor_load_to_lds(g0, g1, gz, gz, 0);
#endif
}
#endif

// ---------------------------------------------------------------- WMMA GEMM
// C[m,n] = sum_k A[m,k] * B[n,k]   (A: MxK bf16 row-major, B: NxK bf16 row-major)
// MODE 0: scatter into q/k/v layouts (qkv projection)
// MODE 2: fp32 out + bias (output projection)
#define BM 128
#define BN 128
#define BK 32
#define LDSW 40   // padded LDS row stride (elements): 80B rows

template <int MODE>
__global__ __launch_bounds__(256) void gemm_bf16_kernel(
    const ushort_t* __restrict__ A, const ushort_t* __restrict__ Bw,
    int M, int N, int K,
    ushort_t* __restrict__ qb, ushort_t* __restrict__ kb, ushort_t* __restrict__ vb,
    float* __restrict__ out, const float* __restrict__ bias)
{
    __shared__ __align__(16) ushort_t As[2][BM * LDSW];
    __shared__ __align__(16) ushort_t Bs[2][BN * LDSW];

    const int tid  = threadIdx.x;
    const int lane = tid & 31;
    const int wv   = tid >> 5;
    const int wm   = wv >> 2;            // 0..1
    const int wn   = wv & 3;             // 0..3
    const int rowbase = wm * 64;
    const int colbase = wn * 32;
    const int half = lane >> 4;
    const int l16  = lane & 15;

    const int m0 = blockIdx.y * BM;
    const int n0 = blockIdx.x * BN;

    const ushort_t* gA = A  + (size_t)m0 * K;
    const ushort_t* gB = Bw + (size_t)n0 * K;

    v8f acc[4][2] = {};
    const int kTiles = K / BK;
    int buf = 0;

#if USE_TDM
    const unsigned ldsA[2] = { (unsigned)(uintptr_t)&As[0][0], (unsigned)(uintptr_t)&As[1][0] };
    const unsigned ldsB[2] = { (unsigned)(uintptr_t)&Bs[0][0], (unsigned)(uintptr_t)&Bs[1][0] };
    if (tid < 32) {                      // wave 0 drives the Tensor Data Mover
        tdm_load_tile(ldsA[0], gA, K);
        tdm_load_tile(ldsB[0], gB, K);
    }
#else
    const int ldr = tid >> 2;            // 0..63
    const int ldc = (tid & 3) * 8;
    uint4 ra0, ra1, rb0, rb1;
    auto g2r = [&](int kt) {
        const int k0 = kt * BK;
        ra0 = *(const uint4*)(gA + (size_t)(ldr)      * K + k0 + ldc);
        ra1 = *(const uint4*)(gA + (size_t)(ldr + 64) * K + k0 + ldc);
        rb0 = *(const uint4*)(gB + (size_t)(ldr)      * K + k0 + ldc);
        rb1 = *(const uint4*)(gB + (size_t)(ldr + 64) * K + k0 + ldc);
    };
    auto r2lds = [&](int b) {
        *(uint4*)&As[b][(ldr)      * LDSW + ldc] = ra0;
        *(uint4*)&As[b][(ldr + 64) * LDSW + ldc] = ra1;
        *(uint4*)&Bs[b][(ldr)      * LDSW + ldc] = rb0;
        *(uint4*)&Bs[b][(ldr + 64) * LDSW + ldc] = rb1;
    };
    g2r(0);
    r2lds(0);
#endif

    for (int kt = 0; kt < kTiles; ++kt) {
        const bool more = (kt + 1) < kTiles;
#if USE_TDM
        if (tid < 32) __builtin_amdgcn_s_wait_tensorcnt((short)0);  // tile kt landed
        __syncthreads();                                            // release compute waves
        if (more && tid < 32) {                                     // DMA next tile over compute
            tdm_load_tile(ldsA[buf ^ 1], gA + (size_t)(kt + 1) * BK, K);
            tdm_load_tile(ldsB[buf ^ 1], gB + (size_t)(kt + 1) * BK, K);
        }
#else
        if (more) g2r(kt + 1);
        __syncthreads();
#endif

        Frag af[4], bf[2];
#pragma unroll
        for (int tm = 0; tm < 4; ++tm) {
            // A 16x32 bf16 layout: lanes<16 rows M, K chunks {0..7},{16..23}; lanes>=16: {8..15},{24..31}
            const int r  = rowbase + tm * 16 + l16;
            const int c0 = 8 * half;
            af[tm].u4[0] = *(const uint4*)&As[buf][r * LDSW + c0];
            af[tm].u4[1] = *(const uint4*)&As[buf][r * LDSW + c0 + 16];
        }
#pragma unroll
        for (int tn = 0; tn < 2; ++tn) {
            // B 32x16 bf16 layout: lanes<16 col N, K chunks {0..7},{16..23}; lanes>=16: {8..15},{24..31}
            const int nn = colbase + tn * 16 + l16;
            const int k0 = 8 * half;
            bf[tn].u4[0] = *(const uint4*)&Bs[buf][nn * LDSW + k0];
            bf[tn].u4[1] = *(const uint4*)&Bs[buf][nn * LDSW + k0 + 16];
        }
#pragma unroll
        for (int tm = 0; tm < 4; ++tm)
#pragma unroll
            for (int tn = 0; tn < 2; ++tn)
                acc[tm][tn] = __builtin_amdgcn_wmma_f32_16x16x32_bf16(
                    false, af[tm].v, false, bf[tn].v, (short)0, acc[tm][tn], false, false);

#if !USE_TDM
        if (more) r2lds(buf ^ 1);
#endif
        buf ^= 1;
    }

    // epilogue: C/D layout lane<16 -> M=v, N=lane; lane>=16 -> M=v+8, N=lane-16
#pragma unroll
    for (int tm = 0; tm < 4; ++tm)
#pragma unroll
        for (int tn = 0; tn < 2; ++tn)
#pragma unroll
            for (int vv = 0; vv < 8; ++vv) {
                const float val = acc[tm][tn][vv];
                const int gm = m0 + rowbase + tm * 16 + vv + 8 * half;
                const int gn = n0 + colbase + tn * 16 + l16;
                if (MODE == 0) {
                    const int b   = gm / (64 * 196);
                    const int rem = gm % (64 * 196);
                    const int t   = rem / 196;
                    const int n   = rem % 196;
                    const int s   = gn / 768;
                    const int hc  = gn % 768;
                    const int h   = hc >> 6;
                    const int d   = hc & 63;
                    const size_t base = ((size_t)((b * 12 + h) * 196 + n)) << 12; // *64*64
                    const ushort_t bv = f32_bf16(val);
                    if (s == 0)      qb[base + t * 64 + d] = bv;     // q[b,h,n,t,d]
                    else if (s == 1) kb[base + t * 64 + d] = bv;     // k[b,h,n,t,d]
                    else             vb[base + d * 64 + t] = bv;     // v[b,h,n,d,t] (transposed)
                } else {
                    out[(size_t)gm * N + gn] = val + bias[gn];
                }
            }
}

// ---------------------------------------------------------------- causal temporal attention
// One workgroup (4 waves) per (b,h,n). Wave i owns q rows 16i..16i+15.
#define PST 72   // P LDS row stride (elements): 144B rows, 16B aligned chunks

__global__ __launch_bounds__(128) void attn_kernel(
    const ushort_t* __restrict__ q, const ushort_t* __restrict__ k,
    const ushort_t* __restrict__ v, ushort_t* __restrict__ attn)
{
    __shared__ __align__(16) ushort_t P[4][16 * PST];

    const int tid  = threadIdx.x;
    const int lane = tid & 31;
    const int i    = tid >> 5;          // wave id == q row-block (wave-uniform)
    const int half = lane >> 4;
    const int l16  = lane & 15;

    const int bhn = blockIdx.x;
    const int b = bhn / (12 * 196);
    const int h = (bhn / 196) % 12;
    const int n = bhn % 196;
    const size_t base = (size_t)bhn * 4096;   // T*hd

    const ushort_t* qp = q + base;
    const ushort_t* kp = k + base;
    const ushort_t* vp = v + base;

    // Q A-fragments for the two K=32 steps over hd=64
    Frag qf[2];
#pragma unroll
    for (int ks = 0; ks < 2; ++ks) {
        const int row = i * 16 + l16;
        const int d0  = 8 * half + 32 * ks;
        qf[ks].u4[0] = *(const uint4*)(qp + row * 64 + d0);
        qf[ks].u4[1] = *(const uint4*)(qp + row * 64 + d0 + 16);
    }

    // S = Q @ K^T, fully unrolled so sacc[] indices are compile-time
    // (wave-uniform predicate: EXEC stays all-ones for the WMMAs).
    v8f sacc[4] = {};
#pragma unroll
    for (int j = 0; j < 4; ++j) {
        if (j <= i) {
#pragma unroll
            for (int ks = 0; ks < 2; ++ks) {
                Frag kf;
                const int tk = j * 16 + l16;
                const int d0 = 8 * half + 32 * ks;
                kf.u4[0] = *(const uint4*)(kp + tk * 64 + d0);
                kf.u4[1] = *(const uint4*)(kp + tk * 64 + d0 + 16);
                sacc[j] = __builtin_amdgcn_wmma_f32_16x16x32_bf16(
                    false, qf[ks].v, false, kf.v, (short)0, sacc[j], false, false);
            }
        }
    }

    // scale + causal mask + safe softmax (row stats live in 16-lane halves)
    float ps[4][8];
#pragma unroll
    for (int j = 0; j < 4; ++j)
#pragma unroll
        for (int vv = 0; vv < 8; ++vv) {
            if (j <= i) {
                float s = sacc[j][vv] * 0.125f;                 // hd^-0.5
                const int Ml = vv + 8 * half;                   // local q row
                if (j == i && l16 > Ml) s = -3.0e38f;           // causal mask on diagonal tile
                ps[j][vv] = s;
            } else {
                ps[j][vv] = -3.0e38f;
            }
        }
    float pr[4][8];
#pragma unroll
    for (int vv = 0; vv < 8; ++vv) {
        float m = -3.0e38f;
#pragma unroll
        for (int j = 0; j < 4; ++j) m = fmaxf(m, ps[j][vv]);
#pragma unroll
        for (int off = 1; off < 16; off <<= 1) m = fmaxf(m, __shfl_xor(m, off, 32));
        float sum = 0.f;
#pragma unroll
        for (int j = 0; j < 4; ++j) {
            const float e = (j <= i) ? __expf(ps[j][vv] - m) : 0.f;
            pr[j][vv] = e;
            sum += e;
        }
#pragma unroll
        for (int off = 1; off < 16; off <<= 1) sum += __shfl_xor(sum, off, 32);
        const float inv = 1.0f / sum;
#pragma unroll
        for (int j = 0; j < 4; ++j) pr[j][vv] *= inv;
    }

    // P -> LDS as bf16 (zeros in masked tiles so the padded K-steps are exact)
#pragma unroll
    for (int j = 0; j < 4; ++j)
#pragma unroll
        for (int vv = 0; vv < 8; ++vv) {
            const int Ml = vv + 8 * half;
            P[i][Ml * PST + j * 16 + l16] = f32_bf16(pr[j][vv]);
        }
    __syncthreads();

    // O = P @ V, unrolled with wave-uniform predicate (ks < i/2+1)
    v8f oacc[4] = {};
#pragma unroll
    for (int ks = 0; ks < 2; ++ks) {
        if (ks <= (i >> 1)) {
            Frag pf;
            const int t0 = 8 * half + 32 * ks;
            pf.u4[0] = *(const uint4*)&P[i][l16 * PST + t0];
            pf.u4[1] = *(const uint4*)&P[i][l16 * PST + t0 + 16];
#pragma unroll
            for (int dt = 0; dt < 4; ++dt) {
                Frag vf;
                const int d  = dt * 16 + l16;
                const int t1 = 8 * half + 32 * ks;
                vf.u4[0] = *(const uint4*)(vp + d * 64 + t1);   // v stored [d][t] -> contiguous
                vf.u4[1] = *(const uint4*)(vp + d * 64 + t1 + 16);
                oacc[dt] = __builtin_amdgcn_wmma_f32_16x16x32_bf16(
                    false, pf.v, false, vf.v, (short)0, oacc[dt], false, false);
            }
        }
    }

    // store to attn buffer in (B,T,N,C) token-major layout (bf16) for the proj GEMM
#pragma unroll
    for (int dt = 0; dt < 4; ++dt)
#pragma unroll
        for (int vv = 0; vv < 8; ++vv) {
            const int tq = i * 16 + vv + 8 * half;
            const int d  = dt * 16 + l16;
            const size_t token = ((size_t)b * 64 + tq) * 196 + n;
            attn[token * 768 + h * 64 + d] = f32_bf16(oacc[dt][vv]);
        }
}

// ---------------------------------------------------------------- launcher
extern "C" void kernel_launch(void* const* d_in, const int* in_sizes, int n_in,
                              void* d_out, int out_size, void* d_ws, size_t ws_size,
                              hipStream_t stream)
{
    const float* x      = (const float*)d_in[0];
    const float* w_qkv  = (const float*)d_in[1];
    const float* w_proj = (const float*)d_in[2];
    const float* b_proj = (const float*)d_in[3];
    float* out = (float*)d_out;

    const int Mtok = 4 * 64 * 196;   // 50176 tokens
    const int C    = 768;
    const int N3   = 3 * C;          // 2304

    char* ws = (char*)d_ws;
    size_t off = 0;
    ushort_t* xb    = (ushort_t*)(ws + off); off += (size_t)Mtok * C * 2;
    ushort_t* wqb   = (ushort_t*)(ws + off); off += (size_t)N3 * C * 2;
    ushort_t* wpb   = (ushort_t*)(ws + off); off += (size_t)C * C * 2;
    ushort_t* qb    = (ushort_t*)(ws + off); off += (size_t)Mtok * C * 2;
    ushort_t* kb    = (ushort_t*)(ws + off); off += (size_t)Mtok * C * 2;
    ushort_t* vb    = (ushort_t*)(ws + off); off += (size_t)Mtok * C * 2;
    ushort_t* attnb = (ushort_t*)(ws + off); off += (size_t)Mtok * C * 2;

    const int nx = Mtok * C;
    cvt_bf16_kernel<<<(nx + 255) / 256, 256, 0, stream>>>(x, xb, nx);
    const int nq = N3 * C;
    cvt_bf16_kernel<<<(nq + 255) / 256, 256, 0, stream>>>(w_qkv, wqb, nq);
    const int np = C * C;
    cvt_bf16_kernel<<<(np + 255) / 256, 256, 0, stream>>>(w_proj, wpb, np);

    dim3 g1(N3 / BN, Mtok / BM);   // 18 x 392
    gemm_bf16_kernel<0><<<g1, 256, 0, stream>>>(xb, wqb, Mtok, N3, C,
                                                qb, kb, vb, nullptr, nullptr);

    attn_kernel<<<4 * 12 * 196, 128, 0, stream>>>(qb, kb, vb, attnb);

    dim3 g2(C / BN, Mtok / BM);    // 6 x 392
    gemm_bf16_kernel<2><<<g2, 256, 0, stream>>>(attnb, wpb, Mtok, C, C,
                                                nullptr, nullptr, nullptr, out, b_proj);
}
